// Model_43490838839356
// MI455X (gfx1250) — matrix operand order repaired and checked by reading, not compiled, not run on hardware
//
#include <hip/hip_runtime.h>

// ---------------------------------------------------------------------------
// LSTM LM forward for MI455X (gfx1250, wave32, WMMA).
// B=128, T=1024, V=256, H=256, 4H=1024.  Gate order i, j, f, o; forget bias 1.
// ---------------------------------------------------------------------------

typedef __attribute__((ext_vector_type(16))) __bf16 v16bf;
typedef __attribute__((ext_vector_type(8)))  float  v8f;

#define B_ 128
#define T_ 1024
#define V_ 256
#define H_ 256
#define G_ 1024   // 4*H
#define K_ 512    // H + V (fused [h|x] GEMM reduction dim)

union AFrag { unsigned int u[8]; v16bf v; };
union CFrag { v8f v; float f[8]; };

__device__ __forceinline__ unsigned short f2bf(float f) {
  unsigned int u = __float_as_uint(f);
  u += 0x7FFFu + ((u >> 16) & 1u);     // round-to-nearest-even
  return (unsigned short)(u >> 16);
}
__device__ __forceinline__ unsigned int pack2bf(float lo, float hi) {
  return (unsigned int)f2bf(lo) | ((unsigned int)f2bf(hi) << 16);
}
__device__ __forceinline__ float sigmoidf_(float x) {
  return 1.0f / (1.0f + __expf(-x));
}

// ---------------------------------------------------------------------------
// Weight prep: W = concat_K(Wh, Wx) -> bf16, transposed to [p][k] with the
// gate-interleaved column permutation p = 4*hcol + gate  (orig col = gate*H+hcol).
// Row-contiguous K so B-operand lanes load 8 consecutive dwords per 32-K block.
// ---------------------------------------------------------------------------
__global__ void prep_WT(const float* __restrict__ Wx, const float* __restrict__ Wh,
                        unsigned int* __restrict__ WT) {
  int idx = blockIdx.x * blockDim.x + threadIdx.x;        // over G_ * K_/2
  if (idx >= G_ * (K_ / 2)) return;
  int p  = idx >> 8;                                      // permuted column
  int kp = idx & 255;                                     // k-pair index
  int gate = p & 3, hcol = p >> 2;
  int co = gate * H_ + hcol;                              // original gate column
  int k0 = 2 * kp, k1 = k0 + 1;
  float a = (k0 < H_) ? Wh[(size_t)k0 * G_ + co] : Wx[(size_t)(k0 - H_) * G_ + co];
  float b = (k1 < H_) ? Wh[(size_t)k1 * G_ + co] : Wx[(size_t)(k1 - H_) * G_ + co];
  WT[idx] = pack2bf(a, b);
}

__global__ void prep_WdT(const float* __restrict__ Wd, unsigned int* __restrict__ WdT) {
  int idx = blockIdx.x * blockDim.x + threadIdx.x;        // over V_ * H_/2
  if (idx >= V_ * (H_ / 2)) return;
  int n  = idx >> 7;                                      // output vocab column
  int kp = idx & 127;
  float a = Wd[(size_t)(2 * kp) * V_ + n];
  float b = Wd[(size_t)(2 * kp + 1) * V_ + n];
  WdT[idx] = pack2bf(a, b);
}

// ---------------------------------------------------------------------------
// Persistent recurrence kernel: 8 blocks x 1024 threads (32 waves).
// Block owns 16 batch rows; wave w owns gate-column tiles w and w+32.
// Per step: WMMA gates = [h|x_t] @ W  (K=512 -> 16 k-blocks, 32 WMMA/wave),
// then in-register LSTM cell update via quad shuffles; h written back to the
// LDS A-tile (bf16) and to hs for the final projection GEMM.
// ---------------------------------------------------------------------------
__global__ __launch_bounds__(1024) void lstm_seq(
    const float* __restrict__ x, const float* __restrict__ bias,
    const unsigned int* __restrict__ WT,
    unsigned short* __restrict__ hs,          // [B*T][H] bf16
    float* __restrict__ out_c, float* __restrict__ out_h) {
  __shared__ unsigned int A[16][K_ / 2];      // [row][k-pair] bf16, 16 KB

  const int tid  = threadIdx.x;
  const int lane = tid & 31;
  const int wave = tid >> 5;
  const int b0   = blockIdx.x * 16;

  const int m   = lane & 15;                  // A row / B column within tile
  const int kh  = lane >> 4;                  // K-half selector (contiguous K order)
  const int nt0 = wave;                       // tiles 0..31
  const int nt1 = wave + 32;                  // tiles 32..63
  const int gate = lane & 3;                  // quad role: i,j,f,o
  const int hq0 = nt0 * 4 + (m >> 2);         // hidden column this quad owns
  const int hq1 = nt1 * 4 + (m >> 2);
  const float bias0 = bias[gate * H_ + hq0];
  const float bias1 = bias[gate * H_ + hq1];

  unsigned short* As = (unsigned short*)&A[0][0];   // halfword view, 512/row

  // h(0) = 0 in the A tile's low-K half; load x_0 into the high-K half.
  for (int i = tid; i < 16 * 128; i += 1024) A[i >> 7][i & 127] = 0u;
  {
    const int r = tid >> 6;
    const int v = (tid & 63) * 4;
    const float* px = x + ((size_t)(b0 + r) * T_ + 0) * V_ + v;
    float4 xv = *(const float4*)px;
    __builtin_prefetch(px + V_, 0, 2);              // global_prefetch_b8: x_{t+1}
    A[r][128 + (v >> 1)]     = pack2bf(xv.x, xv.y);
    A[r][128 + (v >> 1) + 1] = pack2bf(xv.z, xv.w);
  }

  float creg0[8], creg1[8];
  #pragma unroll
  for (int r = 0; r < 8; ++r) { creg0[r] = 0.0f; creg1[r] = 0.0f; }

  for (int t = 0; t < T_; ++t) {
    __syncthreads();                                // A tile complete

    CFrag acc0, acc1;
    #pragma unroll
    for (int i = 0; i < 8; ++i) { acc0.f[i] = 0.0f; acc1.f[i] = 0.0f; }

    #pragma unroll 4
    for (int kb = 0; kb < 16; ++kb) {
      AFrag a;
      #pragma unroll
      for (int j = 0; j < 8; ++j) {                 // contiguous K: k = 16*kh + s
        int k = kh * 16 + 2 * j;
        a.u[j] = A[m][kb * 16 + (k >> 1)];
      }
      AFrag bf0, bf1;
      const unsigned int* w0 = WT + (size_t)(nt0 * 16 + m) * (K_ / 2) + kb * 16 + kh * 8;
      const unsigned int* w1 = WT + (size_t)(nt1 * 16 + m) * (K_ / 2) + kb * 16 + kh * 8;
      #pragma unroll
      for (int j = 0; j < 8; ++j) { bf0.u[j] = w0[j]; bf1.u[j] = w1[j]; }
      acc0.v = __builtin_amdgcn_wmma_f32_16x16x32_bf16(false, a.v, false, bf0.v,
                                                       (short)0, acc0.v, false, false);
      acc1.v = __builtin_amdgcn_wmma_f32_16x16x32_bf16(false, a.v, false, bf1.v,
                                                       (short)0, acc1.v, false, false);
    }

    __syncthreads();                                // all A reads done

    // In-register LSTM cell update. C layout: VGPR r -> row kh*8+r, col = lane&15.
    const int base = lane & ~3;
    const int last = (t == T_ - 1);
    #pragma unroll
    for (int r = 0; r < 8; ++r) {
      const int row = kh * 8 + r;
      // --- tile nt0 ---
      {
        float g = acc0.f[r] + bias0;
        float act = (gate == 1) ? tanhf(g)
                                : sigmoidf_((gate == 2) ? (g + 1.0f) : g);
        float ai = __shfl(act, base + 0, 32);
        float aj = __shfl(act, base + 1, 32);
        float af = __shfl(act, base + 2, 32);
        float ao = __shfl(act, base + 3, 32);
        float c = af * creg0[r] + ai * aj;
        creg0[r] = c;
        float h = ao * tanhf(c);
        if ((lane & 3) == 0) {
          unsigned short hb = f2bf(h);
          As[row * K_ + hq0] = hb;                  // next step's A operand
          hs[((size_t)(b0 + row) * T_ + t) * H_ + hq0] = hb;
          if (last) {
            out_c[(size_t)(b0 + row) * H_ + hq0] = c;
            out_h[(size_t)(b0 + row) * H_ + hq0] = h;
          }
        }
      }
      // --- tile nt1 ---
      {
        float g = acc1.f[r] + bias1;
        float act = (gate == 1) ? tanhf(g)
                                : sigmoidf_((gate == 2) ? (g + 1.0f) : g);
        float ai = __shfl(act, base + 0, 32);
        float aj = __shfl(act, base + 1, 32);
        float af = __shfl(act, base + 2, 32);
        float ao = __shfl(act, base + 3, 32);
        float c = af * creg1[r] + ai * aj;
        creg1[r] = c;
        float h = ao * tanhf(c);
        if ((lane & 3) == 0) {
          unsigned short hb = f2bf(h);
          As[row * K_ + hq1] = hb;
          hs[((size_t)(b0 + row) * T_ + t) * H_ + hq1] = hb;
          if (last) {
            out_c[(size_t)(b0 + row) * H_ + hq1] = c;
            out_h[(size_t)(b0 + row) * H_ + hq1] = h;
          }
        }
      }
    }

    if (t + 1 < T_) {                               // stage x_{t+1}
      const int r = tid >> 6;
      const int v = (tid & 63) * 4;
      const float* px = x + ((size_t)(b0 + r) * T_ + (t + 1)) * V_ + v;
      float4 xv = *(const float4*)px;
      __builtin_prefetch(px + V_, 0, 2);
      A[r][128 + (v >> 1)]     = pack2bf(xv.x, xv.y);
      A[r][128 + (v >> 1) + 1] = pack2bf(xv.z, xv.w);
    }
  }
}

// ---------------------------------------------------------------------------
// logits = hs @ Wd + bd : M=B*T=131072, K=256, N=256. One 16-row M tile per
// block (8 waves); wave w covers N tiles w and w+8. WMMA bf16, fp32 out.
// ---------------------------------------------------------------------------
__global__ __launch_bounds__(256) void logits_gemm(
    const unsigned int* __restrict__ hs, const unsigned int* __restrict__ WdT,
    const float* __restrict__ bd, float* __restrict__ out) {
  const int tid  = threadIdx.x;
  const int lane = tid & 31;
  const int wave = tid >> 5;
  const int m0   = blockIdx.x * 16;
  const int m  = lane & 15;
  const int kh = lane >> 4;
  const int nt0 = wave, nt1 = wave + 8;

  CFrag acc0, acc1;
  #pragma unroll
  for (int i = 0; i < 8; ++i) { acc0.f[i] = 0.0f; acc1.f[i] = 0.0f; }

  #pragma unroll
  for (int kb = 0; kb < 8; ++kb) {
    AFrag a;
    #pragma unroll
    for (int j = 0; j < 8; ++j) {
      int k = kh * 16 + 2 * j;
      a.u[j] = hs[(size_t)(m0 + m) * (H_ / 2) + kb * 16 + (k >> 1)];
    }
    AFrag b0f, b1f;
    const unsigned int* w0 = WdT + (size_t)(nt0 * 16 + m) * (H_ / 2) + kb * 16 + kh * 8;
    const unsigned int* w1 = WdT + (size_t)(nt1 * 16 + m) * (H_ / 2) + kb * 16 + kh * 8;
    #pragma unroll
    for (int j = 0; j < 8; ++j) { b0f.u[j] = w0[j]; b1f.u[j] = w1[j]; }
    acc0.v = __builtin_amdgcn_wmma_f32_16x16x32_bf16(false, a.v, false, b0f.v,
                                                     (short)0, acc0.v, false, false);
    acc1.v = __builtin_amdgcn_wmma_f32_16x16x32_bf16(false, a.v, false, b1f.v,
                                                     (short)0, acc1.v, false, false);
  }

  #pragma unroll
  for (int r = 0; r < 8; ++r) {
    const int row = m0 + kh * 8 + r;
    out[(size_t)row * V_ + nt0 * 16 + m] = acc0.f[r] + bd[nt0 * 16 + m];
    out[(size_t)row * V_ + nt1 * 16 + m] = acc1.f[r] + bd[nt1 * 16 + m];
  }
}

// ---------------------------------------------------------------------------
extern "C" void kernel_launch(void* const* d_in, const int* in_sizes, int n_in,
                              void* d_out, int out_size, void* d_ws, size_t ws_size,
                              hipStream_t stream) {
  (void)in_sizes; (void)n_in; (void)out_size; (void)ws_size;

  const float* x  = (const float*)d_in[0];   // [B,T,V]
  const float* Wx = (const float*)d_in[1];   // [V,4H]
  const float* Wh = (const float*)d_in[2];   // [H,4H]
  const float* b  = (const float*)d_in[3];   // [4H]
  const float* Wd = (const float*)d_in[4];   // [H,V]
  const float* bd = (const float*)d_in[5];   // [V]

  float* out_logits = (float*)d_out;                         // B*T*V
  float* out_c = out_logits + (size_t)B_ * T_ * V_;          // B*H
  float* out_h = out_c + (size_t)B_ * H_;                    // B*H

  // Workspace (~65.2 MB): WT 1 MB, WdT 128 KB, hs 64 MB (all bf16).
  unsigned int* WT   = (unsigned int*)d_ws;
  unsigned int* WdT  = WT  + (size_t)G_ * (K_ / 2);
  unsigned int* hsu  = WdT + (size_t)V_ * (H_ / 2);

  prep_WT <<<dim3(256), dim3(1024), 0, stream>>>(Wx, Wh, WT);
  prep_WdT<<<dim3(32),  dim3(1024), 0, stream>>>(Wd, WdT);
  lstm_seq<<<dim3(B_ / 16), dim3(1024), 0, stream>>>(
      x, b, WT, (unsigned short*)hsu, out_c, out_h);
  logits_gemm<<<dim3((B_ * T_) / 16), dim3(256), 0, stream>>>(
      hsu, WdT, bd, out_logits);
}
